// MultiHeadAttention_5471788335772
// MI455X (gfx1250) — compile-verified, matrix operand order checked by reading, not run on hardware
//
#include <hip/hip_runtime.h>
#include <hip/hip_bf16.h>

// ---------------------------------------------------------------------------
// Multi-head attention for MI455X (gfx1250), wave32, all matmuls through
// v_wmma_f32_16x16x32_f16 (f16 inputs, f32 accumulation), with async
// global->LDS staging (ASYNCcnt) where the layout allows it.
//
// Shapes: B=2, S=2048, d_in=hid=512, H=8, dh=64.
//   1) proj_qkv : Q/K/V = X @ W, 32Mx64N per wave (8 accumulators),
//                 head-major f16 output; Q pre-scaled by 1/sqrt(dh).
//   2) attn     : flash attention, 4-wave workgroups cooperatively staging
//                 shared 32-key K/V tiles; per-wave 16-query tile, online
//                 softmax; mask length = valid_length[bh % B] (np.tile quirk).
//   3) out_proj : out = O_cat @ Wo, 32Mx64N per wave, f32 result.
// ---------------------------------------------------------------------------

#define B_ 2
#define S_ 2048
#define D_ 512
#define H_ 8
#define DH_ 64
#define BH_ (B_ * H_)
#define MROWS_ (B_ * S_)          // 4096
#define MASK_VALUE -1000000.0f

typedef __attribute__((ext_vector_type(16))) _Float16 v16h;
typedef __attribute__((ext_vector_type(8)))  float    v8f;
typedef int av4i __attribute__((vector_size(16)));   // builtin operand type

#define USE_ASYNC_LDS 0
#if defined(__has_builtin)
#if __has_builtin(__builtin_amdgcn_global_load_async_to_lds_b128) && \
    __has_builtin(__builtin_amdgcn_s_wait_asynccnt)
#undef USE_ASYNC_LDS
#define USE_ASYNC_LDS 1
#endif
#endif

// Copy 8 halfs (16B) global -> LDS. Async path uses the gfx1250
// GLOBAL_LOAD_ASYNC_TO_LDS_B128 instruction (tracked by ASYNCcnt).
// The builtin expects (v4i addrspace(1)*, v4i addrspace(3)*, imm, imm cpol).
__device__ __forceinline__ void copy_b128_to_lds(const _Float16* g,
                                                 _Float16* l) {
#if USE_ASYNC_LDS
  __builtin_amdgcn_global_load_async_to_lds_b128(
      (__attribute__((address_space(1))) av4i*)(g),
      (__attribute__((address_space(3))) av4i*)(l), 0, 0);
#else
  *(uint4*)l = *(const uint4*)g;
#endif
}

__device__ __forceinline__ void wait_async_lds() {
#if USE_ASYNC_LDS
  __builtin_amdgcn_s_wait_asynccnt(0);
#endif
}

// K-index of element pair i (i in 0..7) for a lane in half-group `half`,
// per the CDNA5 16-bit A/B operand VGPR layout (CDNA5 ISA):
//   V0..V3 hold K = half*8 + {0..7}, V4..V7 hold K = 16 + half*8 + {0..7}.
__device__ __forceinline__ int kpat(int i, int half) {
  return ((i & 4) << 2) + (half << 3) + ((i & 3) << 1);
}

// Gather one 16-half WMMA operand from an LDS tile laid out [outer][K]
// with `stride` halfs per outer row. Pairs (K, K+1) are consecutive, so the
// compiler merges each pair into a 32-bit (or 2addr) ds_load.
__device__ __forceinline__ v16h lds_gather(const _Float16* base, int outer,
                                           int stride, int koff, int half) {
  v16h r;
#pragma unroll
  for (int i = 0; i < 8; ++i) {
    const int K0 = koff + kpat(i, half);
    union { unsigned int u; _Float16 h[2]; } cv;
    cv.u = *(const unsigned int*)(base + outer * stride + K0);
    r[2 * i]     = cv.h[0];
    r[2 * i + 1] = cv.h[1];
  }
  return r;
}

__device__ __forceinline__ v8f wmma_f16(v16h a, v16h b, v8f c) {
  return __builtin_amdgcn_wmma_f32_16x16x32_f16(false, a, false, b,
                                                (short)0, c, false, false);
}

// ---------------------------------------------------------------------------
// Kernel 1: QKV projection.  grid = (N/64, M/32, 3), block = 32 (one wave).
// Each wave computes a 32x64 output block (2 m-tiles x 4 n-tiles, 8 WMMAs
// per 32-deep k-step). A: (4096 x 512) f32, W: (512 x 512) f32.
// Output written head-major f16: O[(b*H + h)*S*DH + s*DH + d].
// ---------------------------------------------------------------------------
__global__ __launch_bounds__(32)
void proj_qkv(const float* __restrict__ q, const float* __restrict__ k,
              const float* __restrict__ v,
              const float* __restrict__ Wq, const float* __restrict__ Wk,
              const float* __restrict__ Wv,
              _Float16* __restrict__ Qh, _Float16* __restrict__ Kh,
              _Float16* __restrict__ Vh) {
  const int which = blockIdx.z;
  const float* A = (which == 0) ? q : (which == 1) ? k : v;
  const float* W = (which == 0) ? Wq : (which == 1) ? Wk : Wv;
  _Float16* O    = (which == 0) ? Qh : (which == 1) ? Kh : Vh;
  const float scale = (which == 0) ? 0.125f : 1.0f;  // 1/sqrt(dh) into Q

  const int lane = threadIdx.x;
  const int half = lane >> 4;
  const int lo   = lane & 15;
  const int nbase = blockIdx.x * 64;
  const int mbase = blockIdx.y * 32;

  __shared__ __align__(16) _Float16 aT[32 * 34];  // [m][k], stride 34
  __shared__ __align__(16) _Float16 bT[64 * 34];  // [n][k]

  v8f c[2][4] = {{{}, {}, {}, {}}, {{}, {}, {}, {}}};

  for (int kb = 0; kb < D_; kb += 32) {
    // Stage A (32x32, f32->f16): one row per iteration, lanes = consecutive k.
#pragma unroll
    for (int i = 0; i < 32; ++i)
      aT[i * 34 + lane] = (_Float16)A[(size_t)(mbase + i) * D_ + kb + lane];
    // Stage W (32k x 64n) transposed into [n][k]; lanes read consecutive n.
#pragma unroll
    for (int i = 0; i < 64; ++i) {
      const int kk = i >> 1;
      const int n = ((i & 1) << 5) + lane;
      bT[n * 34 + kk] = (_Float16)W[(size_t)(kb + kk) * D_ + nbase + n];
    }
    const v16h a0 = lds_gather(aT, lo, 34, 0, half);
    const v16h a1 = lds_gather(aT, 16 + lo, 34, 0, half);
#pragma unroll
    for (int t = 0; t < 4; ++t) {
      const v16h b = lds_gather(bT, t * 16 + lo, 34, 0, half);
      c[0][t] = wmma_f16(a0, b, c[0][t]);
      c[1][t] = wmma_f16(a1, b, c[1][t]);
    }
  }

  // C/D layout: lane -> col (N), VGPR r -> row = r + 8*half.
#pragma unroll
  for (int mi = 0; mi < 2; ++mi)
#pragma unroll
    for (int t = 0; t < 4; ++t)
#pragma unroll
      for (int r = 0; r < 8; ++r) {
        const int row = mbase + mi * 16 + r + 8 * half;
        const int col = nbase + t * 16 + lo;
        const int bb = row >> 11, s = row & (S_ - 1);
        const int h = col >> 6, d = col & (DH_ - 1);
        O[((size_t)(bb * H_ + h) * S_ + s) * DH_ + d] =
            (_Float16)(c[mi][t][r] * scale);
      }
}

// ---------------------------------------------------------------------------
// Kernel 2: flash attention.  grid = (S/64, B*H), block = 128 (4 waves).
// The 4 waves cooperatively stage shared 32-key K/V tiles; each wave owns a
// 16-query tile and keeps its own online-softmax state and accumulators.
// ---------------------------------------------------------------------------
__global__ __launch_bounds__(128)
void attn(const _Float16* __restrict__ Qh, const _Float16* __restrict__ Kh,
          const _Float16* __restrict__ Vh, const int* __restrict__ vlen,
          _Float16* __restrict__ Oh) {
  const int tid  = threadIdx.x;
  const int w    = tid >> 5;        // wave id 0..3
  const int lane = tid & 31;
  const int half = lane >> 4;
  const int lo   = lane & 15;
  const int qbase = (blockIdx.x * 4 + w) * 16;
  const int bh    = blockIdx.y;
  const int vl    = vlen[bh % B_];  // faithful to np.tile(valid_length, H)

  // Strides: 72 halfs = 144B rows (16B aligned for async B128, bank-spread),
  // 36 halfs for the key-indexed tiles.
  __shared__ __align__(16) _Float16 qT[4][16 * 72];  // per-wave [m][dh]
  __shared__ __align__(16) _Float16 kT[32 * 72];     // shared [key][dh]
  __shared__ __align__(16) _Float16 vT[64 * 36];     // shared [dh][key]
  __shared__ __align__(16) _Float16 pT[4][16 * 36];  // per-wave [m][key]

  const _Float16* Qp = Qh + (size_t)bh * S_ * DH_;
  const _Float16* Kp = Kh + (size_t)bh * S_ * DH_;
  const _Float16* Vp = Vh + (size_t)bh * S_ * DH_;
  _Float16* qTw = qT[w];
  _Float16* pTw = pT[w];

  // Stage this wave's Q tile (16 x 64 halfs = 128 x B128), async if available.
#pragma unroll
  for (int i = 0; i < 4; ++i) {
    const int flat = i * 32 + lane;
    const int m = flat >> 3, c8 = (flat & 7) * 8;
    copy_b128_to_lds(Qp + (size_t)(qbase + m) * DH_ + c8, &qTw[m * 72 + c8]);
  }
  wait_async_lds();

  v8f acc[4] = {{}, {}, {}, {}};
  float m_run[8], l_run[8];
#pragma unroll
  for (int r = 0; r < 8; ++r) { m_run[r] = -INFINITY; l_run[r] = 0.0f; }

  for (int kb = 0; kb < S_; kb += 32) {
    __syncthreads();  // all waves done reading previous K/V tiles

    // Cooperative stage of K block row-major (256 x B128 over 128 threads).
#pragma unroll
    for (int i = 0; i < 2; ++i) {
      const int flat = i * 128 + tid;
      const int key = flat >> 3, c8 = (flat & 7) * 8;
      copy_b128_to_lds(Kp + (size_t)(kb + key) * DH_ + c8,
                       &kT[key * 72 + c8]);
    }
    // Cooperative stage of V block transposed [dh][key] (element transpose,
    // read coalesced 32-bit, scatter 16-bit).
    {
      const unsigned int* src = (const unsigned int*)(Vp + (size_t)kb * DH_);
#pragma unroll
      for (int i = 0; i < 8; ++i) {
        const int flat = i * 128 + tid;          // 1024 half-pairs
        const int key = flat >> 5, dp = (flat & 31) * 2;
        union { unsigned int u; _Float16 h[2]; } cv;
        cv.u = src[flat];
        vT[dp * 36 + key]       = cv.h[0];
        vT[(dp + 1) * 36 + key] = cv.h[1];
      }
    }
    wait_async_lds();
    __syncthreads();  // staged tiles visible to all waves

    // Scores: two 16x16 tiles (keys kb+lo, kb+16+lo), K-depth 64 = 2 steps.
    v8f s0 = {}, s1 = {};
#pragma unroll
    for (int ks = 0; ks < 2; ++ks) {
      const v16h aq = lds_gather(qTw, lo, 72, ks * 32, half);
      const v16h b0 = lds_gather(kT, lo, 72, ks * 32, half);
      const v16h b1 = lds_gather(kT, 16 + lo, 72, ks * 32, half);
      s0 = wmma_f16(aq, b0, s0);
      s1 = wmma_f16(aq, b1, s1);
    }

    const bool mask0 = (kb + lo) >= vl;
    const bool mask1 = (kb + 16 + lo) >= vl;

    // Online softmax per query row (row = r + 8*half; cols across 16 lanes).
#pragma unroll
    for (int r = 0; r < 8; ++r) {
      const float x0 = mask0 ? MASK_VALUE : s0[r];
      const float x1 = mask1 ? MASK_VALUE : s1[r];
      float mx = fmaxf(x0, x1);
#pragma unroll
      for (int off = 1; off < 16; off <<= 1)
        mx = fmaxf(mx, __shfl_xor(mx, off, 16));
      const float mn = fmaxf(m_run[r], mx);
      const float rescale = __expf(m_run[r] - mn);   // exp(-inf)=0 first time
      m_run[r] = mn;
      const float p0 = __expf(x0 - mn);
      const float p1 = __expf(x1 - mn);
      float rs = p0 + p1;
#pragma unroll
      for (int off = 1; off < 16; off <<= 1)
        rs += __shfl_xor(rs, off, 16);
      l_run[r] = l_run[r] * rescale + rs;
#pragma unroll
      for (int t = 0; t < 4; ++t) acc[t][r] *= rescale;
      const int prow = r + 8 * half;
      pTw[prow * 36 + lo]      = (_Float16)p0;
      pTw[prow * 36 + 16 + lo] = (_Float16)p1;
    }

    // P (16x32) @ V (32x64): 4 output dh-tiles, one WMMA each.
    const v16h ap = lds_gather(pTw, lo, 36, 0, half);
#pragma unroll
    for (int t = 0; t < 4; ++t) {
      const v16h bv = lds_gather(vT, t * 16 + lo, 36, 0, half);
      acc[t] = wmma_f16(ap, bv, acc[t]);
    }
  }

  // Normalize and store head-major f16 output.
#pragma unroll
  for (int t = 0; t < 4; ++t)
#pragma unroll
    for (int r = 0; r < 8; ++r) {
      const int row = qbase + r + 8 * half;
      Oh[((size_t)bh * S_ + row) * DH_ + t * 16 + lo] =
          (_Float16)(acc[t][r] / l_run[r]);
    }
}

// ---------------------------------------------------------------------------
// Kernel 3: output projection.  grid = (N/64, M/32), block = 32 (one wave).
// out[m][n] = sum_k Ocat[m][k] * Wo[k][n], Ocat read from head-major f16 Oh.
// ---------------------------------------------------------------------------
__global__ __launch_bounds__(32)
void out_proj(const _Float16* __restrict__ Oh, const float* __restrict__ Wo,
              float* __restrict__ out) {
  const int lane = threadIdx.x;
  const int half = lane >> 4;
  const int lo   = lane & 15;
  const int nbase = blockIdx.x * 64;
  const int mbase = blockIdx.y * 32;

  __shared__ __align__(16) _Float16 aT[32 * 34];
  __shared__ __align__(16) _Float16 bT[64 * 34];

  v8f c[2][4] = {{{}, {}, {}, {}}, {{}, {}, {}, {}}};

  for (int kb = 0; kb < D_; kb += 32) {
    const int h = kb >> 6, d0 = kb & (DH_ - 1);  // 32-chunk stays in one head
#pragma unroll
    for (int i = 0; i < 32; ++i) {
      const int row = mbase + i;
      const int bb = row >> 11, s = row & (S_ - 1);
      aT[i * 34 + lane] =
          Oh[((size_t)(bb * H_ + h) * S_ + s) * DH_ + d0 + lane];
    }
#pragma unroll
    for (int i = 0; i < 64; ++i) {
      const int kk = i >> 1;
      const int n = ((i & 1) << 5) + lane;
      bT[n * 34 + kk] = (_Float16)Wo[(size_t)(kb + kk) * D_ + nbase + n];
    }
    const v16h a0 = lds_gather(aT, lo, 34, 0, half);
    const v16h a1 = lds_gather(aT, 16 + lo, 34, 0, half);
#pragma unroll
    for (int t = 0; t < 4; ++t) {
      const v16h b = lds_gather(bT, t * 16 + lo, 34, 0, half);
      c[0][t] = wmma_f16(a0, b, c[0][t]);
      c[1][t] = wmma_f16(a1, b, c[1][t]);
    }
  }

#pragma unroll
  for (int mi = 0; mi < 2; ++mi)
#pragma unroll
    for (int t = 0; t < 4; ++t)
#pragma unroll
      for (int r = 0; r < 8; ++r) {
        const int row = mbase + mi * 16 + r + 8 * half;
        out[(size_t)row * D_ + nbase + t * 16 + lo] = c[mi][t][r];
      }
}

// ---------------------------------------------------------------------------
extern "C" void kernel_launch(void* const* d_in, const int* in_sizes, int n_in,
                              void* d_out, int out_size, void* d_ws,
                              size_t ws_size, hipStream_t stream) {
  const float* query = (const float*)d_in[0];
  const float* key   = (const float*)d_in[1];
  const float* value = (const float*)d_in[2];
  const float* Wq    = (const float*)d_in[3];
  const float* Wk    = (const float*)d_in[4];
  const float* Wv    = (const float*)d_in[5];
  const float* Wo    = (const float*)d_in[6];
  const int*   vlen  = (const int*)d_in[7];

  // Workspace: 4 head-major f16 tensors of BH*S*DH elements -> 16 MB total.
  const size_t tensor_elems = (size_t)BH_ * S_ * DH_;
  _Float16* Qh = (_Float16*)d_ws;
  _Float16* Kh = Qh + tensor_elems;
  _Float16* Vh = Kh + tensor_elems;
  _Float16* Oh = Vh + tensor_elems;

  {
    dim3 grid(D_ / 64, MROWS_ / 32, 3);
    proj_qkv<<<grid, 32, 0, stream>>>(query, key, value, Wq, Wk, Wv,
                                      Qh, Kh, Vh);
  }
  {
    dim3 grid(S_ / 64, BH_);
    attn<<<grid, 128, 0, stream>>>(Qh, Kh, Vh, vlen, Oh);
  }
  {
    dim3 grid(D_ / 64, MROWS_ / 32);
    out_proj<<<grid, 32, 0, stream>>>(Oh, Wo, (float*)d_out);
  }
}
